// SimpleESN_4896262717804
// MI455X (gfx1250) — matrix operand order repaired and checked by reading, not compiled, not run on hardware
//
#include <hip/hip_runtime.h>
#include <hip/hip_bf16.h>

// ---------------------------------------------------------------------------
// ESN recurrence on MI455X (gfx1250), wave32 + WMMA bf16 + LDS-staged A.
//
// B=32, T=2048, F=64, M=2048, P=1, LEAKY=0.9
//   h_{t+1} = 0.1*h_t + 0.9*tanh(x_t @ K + b + h_t @ W)
//   y = h_T @ w_out + b_out
//
// Persistent kernel: 32 blocks x 256 threads = 256 wave32s; one 16x16 state
// tile per wave. A operand (H slab) staged in LDS per block per step
// (16B row pad -> conflict-free ds_load_b128). B operand (packed W fragments)
// streams from L2 with ping-pong register buffers (no copies): loads for the
// next 4 K-chunks land in one buffer while WMMAs consume the other. Two
// independent accumulator chains hide XDL latency. Branch-free tanh via
// v_exp/v_rcp. Grid-wide sync via monotonic atomic counter.
// ---------------------------------------------------------------------------

#define B_ 32
#define T_ 2048
#define F_ 64
#define M_ 2048
#define LEAKY 0.9f

// LDS layout (halfs): padded H slab + padded x slab
#define HROWP 2056            // 2048 + 8 halfs pad  -> row stride 4112 B
#define XROWP 72              // 64 + 8 halfs pad
#define LDS_HALFS (16 * HROWP + 16 * XROWP)
#define LDS_BYTES (LDS_HALFS * 2)   // 68096 B (within 320 KB WGP LDS)

typedef __attribute__((ext_vector_type(16))) __bf16 v16bf;
typedef __attribute__((ext_vector_type(8)))  __bf16 v8bf;
typedef __attribute__((ext_vector_type(8)))  float  v8f;

// ---- workspace layout (bytes) ----
static const size_t WB_OFF  = 0;                                     // 8 MiB
static const size_t KB_OFF  = WB_OFF + (size_t)128 * 64 * 1024;      // 8388608
static const size_t XB_OFF  = KB_OFF + (size_t)128 * 2 * 1024;       // 8650752
static const size_t HF_OFF  = XB_OFF + (size_t)T_ * B_ * F_ * 2;     // 17039360
static const size_t HS_OFF  = HF_OFF + (size_t)2 * B_ * M_ * 4;      // 17563648
static const size_t CTR_OFF = HS_OFF + (size_t)2 * B_ * M_ * 2;      // 17825792

// ---------------------------------------------------------------------------
// Pack W [2048 x 2048] fp32 row-major (W[k][n]) into WMMA-B bf16 fragments.
// Fragment (nt, kc): 512 halfs; lane l holds 16 contiguous halfs:
//   n = nt*16 + (l&15);  k = kc*32 + (l<16 ? 0 : 8) + e + (e<8 ? 0 : 8),  e = 0..15
__global__ void pack_W(const float* __restrict__ W, __bf16* __restrict__ Wb) {
    int gid  = blockIdx.x * blockDim.x + threadIdx.x;
    int lane = gid & 31;
    int frag = gid >> 5;                 // nt*64 + kc
    if (frag >= 128 * 64) return;
    int nt = frag >> 6, kc = frag & 63;
    int n     = nt * 16 + (lane & 15);
    int kbase = kc * 32 + ((lane < 16) ? 0 : 8);
    __bf16* dst = Wb + (size_t)frag * 512 + lane * 16;
#pragma unroll
    for (int e = 0; e < 16; ++e)
        dst[e] = (__bf16)W[(size_t)(kbase + e + (e < 8 ? 0 : 8)) * M_ + n];
}

// Same packing for the input kernel [64 x 2048] (only kc = 0,1).
__global__ void pack_K(const float* __restrict__ Kin, __bf16* __restrict__ Kb) {
    int gid  = blockIdx.x * blockDim.x + threadIdx.x;
    int lane = gid & 31;
    int frag = gid >> 5;                 // nt*2 + kc
    if (frag >= 128 * 2) return;
    int nt = frag >> 1, kc = frag & 1;
    int n     = nt * 16 + (lane & 15);
    int kbase = kc * 32 + ((lane < 16) ? 0 : 8);
    __bf16* dst = Kb + (size_t)frag * 512 + lane * 16;
#pragma unroll
    for (int e = 0; e < 16; ++e)
        dst[e] = (__bf16)Kin[(size_t)(kbase + e + (e < 8 ? 0 : 8)) * M_ + n];
}

// x [B][T][F] fp32 -> Xb [T][B][F] bf16 (A-operand rows contiguous per step).
__global__ void pack_X(const float* __restrict__ x, __bf16* __restrict__ Xb) {
    size_t i = (size_t)blockIdx.x * blockDim.x + threadIdx.x;
    if (i >= (size_t)B_ * T_ * F_) return;
    int f  = (int)(i % F_);
    size_t bt = i / F_;
    int t  = (int)(bt % T_);
    int b  = (int)(bt / T_);
    Xb[((size_t)t * B_ + b) * F_ + f] = (__bf16)x[i];
}

// Zero both state buffers + barrier counter.
__global__ void init_state(float* __restrict__ Hf, __bf16* __restrict__ Hs,
                           unsigned* __restrict__ ctr) {
    int i = blockIdx.x * blockDim.x + threadIdx.x;
    if (i < 2 * B_ * M_) { Hf[i] = 0.0f; Hs[i] = (__bf16)0.0f; }
    if (i == 0) *ctr = 0u;
}

__device__ __forceinline__ v16bf make_a(const __bf16* p) {
    v8bf a0 = *(const v8bf*)(p);
    v8bf a1 = *(const v8bf*)(p + 16);
    return __builtin_shufflevector(a0, a1, 0, 1, 2, 3, 4, 5, 6, 7,
                                   8, 9, 10, 11, 12, 13, 14, 15);
}

// Branch-free tanh: 1 - 2/(exp2(2*log2e*z)+1). Saturates to +/-1, no exec
// masking (v_exp_f32 + v_rcp_f32 TRANS ops only).
__device__ __forceinline__ float fast_tanh(float z) {
    float e = __builtin_amdgcn_exp2f(z * 2.8853900817779268f);
    return 1.0f - 2.0f * __builtin_amdgcn_rcpf(e + 1.0f);
}

// ---------------------------------------------------------------------------
__global__ void __launch_bounds__(256)
esn_scan(const __bf16* __restrict__ Wb, const __bf16* __restrict__ Kb,
         const __bf16* __restrict__ Xb, const float* __restrict__ bias,
         float* __restrict__ Hf, __bf16* __restrict__ Hs,
         unsigned* __restrict__ ctr) {
    extern __shared__ char smem_raw[];
    __bf16* HSH = (__bf16*)smem_raw;          // [16][HROWP]
    __bf16* XSH = HSH + 16 * HROWP;           // [16][XROWP]

    const int lane    = threadIdx.x & 31;
    const int wave    = threadIdx.x >> 5;
    const int tile    = blockIdx.x * 8 + wave;   // 0..255
    const int mt      = tile >> 7;               // 0..1 (uniform per block)
    const int nt      = tile & 127;              // 0..127
    const int halfSel = (lane < 16) ? 0 : 8;     // A-frag K sub-offset
    const int arowL   = lane & 15;               // A-frag row (local to slab)
    const int col     = nt * 16 + (lane & 15);   // C-frag column
    const int rowC    = mt * 16 + ((lane < 16) ? 0 : 8); // C-frag row base

    const float bias_v = bias[col];
    const __bf16* wtile = Wb + (size_t)nt * 64 * 512 + lane * 16;
    const __bf16* ktile = Kb + (size_t)nt * 2 * 512 + lane * 16;

    for (int t = 0; t < T_; ++t) {
        const int cur = t & 1, nxt = cur ^ 1;
        const __bf16* __restrict__ Hc  = Hs + (size_t)cur * B_ * M_;
        const float*  __restrict__ Hfc = Hf + (size_t)cur * B_ * M_;
        float*  __restrict__ Hfn = Hf + (size_t)nxt * B_ * M_;
        __bf16* __restrict__ Hsn = Hs + (size_t)nxt * B_ * M_;

        // ---- stage A operands into LDS (once per block per step) ----
        // H slab: 16 rows x 2048 halfs = 4096 chunks of 16B; 16 per thread.
        {
            const __bf16* src = Hc + (size_t)mt * 16 * M_;
#pragma unroll
            for (int j = 0; j < 16; ++j) {
                const int chunk = threadIdx.x + j * 256;   // 0..4095
                const int row   = chunk >> 8;              // 0..15
                const int c8    = (chunk & 255) * 8;       // half offset in row
                const uint4 v = *(const uint4*)(src + (size_t)row * M_ + c8);
                *(uint4*)(HSH + row * HROWP + c8) = v;
            }
            // x slab: 16 rows x 64 halfs = 128 chunks of 16B.
            if (threadIdx.x < 128) {
                const int row = threadIdx.x >> 3;
                const int c8  = (threadIdx.x & 7) * 8;
                const uint4 v = *(const uint4*)(
                    Xb + ((size_t)t * B_ + mt * 16 + row) * F_ + c8);
                *(uint4*)(XSH + row * XROWP + c8) = v;
            }
        }
        if (t + 1 < T_)   // pull next step's x slab toward the caches early
            __builtin_prefetch(
                Xb + ((size_t)(t + 1) * B_ + mt * 16 + arowL) * F_, 0, 1);
        __syncthreads();

        // Two independent accumulator chains (even/odd K-chunks) to double
        // WMMA ILP; summed after the K loop.
        v8f acc0 = {}, acc1 = {};

        // ---- input projection: x_t(16x64) @ K(64x16), one chunk per chain
        const __bf16* xrowL = XSH + arowL * XROWP;
        {
            v16bf b0 = *(const v16bf*)(ktile);
            v16bf b1 = *(const v16bf*)(ktile + 512);
            v16bf a0 = make_a(xrowL + halfSel);
            v16bf a1 = make_a(xrowL + 32 + halfSel);
            acc0 = __builtin_amdgcn_wmma_f32_16x16x32_bf16(
                false, a0, false, b0, (short)0, acc0, false, false);
            acc1 = __builtin_amdgcn_wmma_f32_16x16x32_bf16(
                false, a1, false, b1, (short)0, acc1, false, false);
        }

        // ---- recurrence: h_t(16x2048) @ W(2048x16), 64 K-chunks ----
        // Ping-pong B buffers (no register copies): loads for the next 4
        // chunks land in one buffer while WMMAs consume the other.
        const __bf16* hrowL = HSH + arowL * HROWP;
        v16bf bb0[4], bb1[4];
#pragma unroll
        for (int i = 0; i < 4; ++i)
            bb0[i] = *(const v16bf*)(wtile + (size_t)i * 512);
#pragma unroll 1
        for (int kb = 0; kb < 64; kb += 8) {
#pragma unroll
            for (int i = 0; i < 4; ++i)      // kb+4+i <= 63: no wrap needed
                bb1[i] = *(const v16bf*)(wtile + (size_t)(kb + 4 + i) * 512);
#pragma unroll
            for (int i = 0; i < 4; ++i) {
                v16bf a = make_a(hrowL + (kb + i) * 32 + halfSel);
                if (i & 1)
                    acc1 = __builtin_amdgcn_wmma_f32_16x16x32_bf16(
                        false, a, false, bb0[i], (short)0, acc1, false, false);
                else
                    acc0 = __builtin_amdgcn_wmma_f32_16x16x32_bf16(
                        false, a, false, bb0[i], (short)0, acc0, false, false);
            }
#pragma unroll
            for (int i = 0; i < 4; ++i)      // wrap (&63): branch-free tail
                bb0[i] = *(const v16bf*)(wtile + (size_t)((kb + 8 + i) & 63) * 512);
#pragma unroll
            for (int i = 0; i < 4; ++i) {
                v16bf a = make_a(hrowL + (kb + 4 + i) * 32 + halfSel);
                if (i & 1)
                    acc1 = __builtin_amdgcn_wmma_f32_16x16x32_bf16(
                        false, a, false, bb1[i], (short)0, acc1, false, false);
                else
                    acc0 = __builtin_amdgcn_wmma_f32_16x16x32_bf16(
                        false, a, false, bb1[i], (short)0, acc0, false, false);
            }
        }

        const v8f acc = acc0 + acc1;

        // ---- leaky tanh update; write fp32 + bf16 next-state buffers ----
#pragma unroll
        for (int r = 0; r < 8; ++r) {
            const int row = rowC + r;
            const size_t idx = (size_t)row * M_ + col;
            const float hp = Hfc[idx];
            const float hn =
                hp * (1.0f - LEAKY) + fast_tanh(acc[r] + bias_v) * LEAKY;
            Hfn[idx] = hn;
            Hsn[idx] = (__bf16)hn;
        }

        // ---- grid-wide barrier: release add by leader, acquire spin by all
        __threadfence();
        __syncthreads();
        if (threadIdx.x == 0)
            __hip_atomic_fetch_add(ctr, 1u, __ATOMIC_RELEASE,
                                   __HIP_MEMORY_SCOPE_AGENT);
        const unsigned target = (unsigned)(t + 1) * 32u;
        while (__hip_atomic_load(ctr, __ATOMIC_ACQUIRE,
                                 __HIP_MEMORY_SCOPE_AGENT) < target)
            __builtin_amdgcn_s_sleep(1);
        __syncthreads();
    }
}

// Readout: y[b] = h_T[b,:] . w + b0  (final state is fp32 buffer 0, T even).
__global__ void readout(const float* __restrict__ Hf,
                        const float* __restrict__ w,
                        const float* __restrict__ b0,
                        float* __restrict__ out) {
    __shared__ float red[256];
    const int b = blockIdx.x;
    float s = 0.0f;
    for (int m = threadIdx.x; m < M_; m += 256)
        s += Hf[(size_t)b * M_ + m] * w[m];
    red[threadIdx.x] = s;
    __syncthreads();
    for (int off = 128; off > 0; off >>= 1) {
        if (threadIdx.x < off) red[threadIdx.x] += red[threadIdx.x + off];
        __syncthreads();
    }
    if (threadIdx.x == 0) out[b] = red[0] + b0[0];
}

extern "C" void kernel_launch(void* const* d_in, const int* in_sizes, int n_in,
                              void* d_out, int out_size, void* d_ws,
                              size_t ws_size, hipStream_t stream) {
    (void)in_sizes; (void)n_in; (void)out_size; (void)ws_size;
    const float* x    = (const float*)d_in[0];  // [32,2048,64]
    const float* kin  = (const float*)d_in[1];  // [64,2048]
    const float* wrec = (const float*)d_in[2];  // [2048,2048]
    const float* bias = (const float*)d_in[3];  // [2048]
    const float* wout = (const float*)d_in[4];  // [2048,1]
    const float* bout = (const float*)d_in[5];  // [1]
    float* out = (float*)d_out;                 // [32,1]

    char* ws = (char*)d_ws;
    __bf16*   Wb  = (__bf16*)(ws + WB_OFF);
    __bf16*   Kb  = (__bf16*)(ws + KB_OFF);
    __bf16*   Xb  = (__bf16*)(ws + XB_OFF);
    float*    Hf  = (float*)(ws + HF_OFF);
    __bf16*   Hs  = (__bf16*)(ws + HS_OFF);
    unsigned* ctr = (unsigned*)(ws + CTR_OFF);

    pack_W<<<(128 * 64 * 32 + 255) / 256, 256, 0, stream>>>(wrec, Wb);
    pack_K<<<(128 * 2 * 32 + 255) / 256, 256, 0, stream>>>(kin, Kb);
    pack_X<<<((size_t)B_ * T_ * F_ + 255) / 256, 256, 0, stream>>>(x, Xb);
    init_state<<<(2 * B_ * M_ + 255) / 256, 256, 0, stream>>>(Hf, Hs, ctr);
    esn_scan<<<32, 256, LDS_BYTES, stream>>>(Wb, Kb, Xb, bias, Hf, Hs, ctr);
    readout<<<B_, 256, 0, stream>>>(Hf, wout, bout, out);
}
